// upAttention_49520972923035
// MI455X (gfx1250) — matrix operand order repaired and checked by reading, not compiled, not run on hardware
//
#include <hip/hip_runtime.h>

// ---------------------------------------------------------------------------
// CDNA5 (gfx1250) wave32 WMMA types
// ---------------------------------------------------------------------------
typedef __attribute__((ext_vector_type(16))) __bf16 v16bf;
typedef __attribute__((ext_vector_type(8)))  float  v8f;

union Frag16 { v16bf v; uint4 q[2]; unsigned short us[16]; };
union FragC  { v8f  v; float f[8]; };
union Pack8  { unsigned short us[8]; uint4 q; };

__device__ __forceinline__ unsigned short f2bf(float x) {
  union { __bf16 h; unsigned short u; } c;
  c.h = (__bf16)x;                 // native f32->bf16 convert
  return c.u;
}
__device__ __forceinline__ float bf2f(unsigned short u) {
  union { unsigned int u; float f; } c;
  c.u = ((unsigned int)u) << 16;   // exact
  return c.f;
}

#define WMMA_BF16(A, B, C) \
  __builtin_amdgcn_wmma_f32_16x16x32_bf16(false, (A), false, (B), (short)0, (C), false, false)

// ---------------------------------------------------------------------------
// small helpers
// ---------------------------------------------------------------------------
__global__ void init_consts(float* ones, float* zeros) {
  int i = threadIdx.x; ones[i] = 1.f; zeros[i] = 0.f;
}

__global__ void cvt_f32_bf16(const float* __restrict__ s, unsigned short* __restrict__ d, int n) {
  int i = blockIdx.x * 256 + threadIdx.x;
  if (i < n) d[i] = f2bf(s[i]);
}

// per-channel batch-norm stats on f32 NCHW -> folded scale/shift: y = x*sA[c]+sB[c]
__global__ void __launch_bounds__(256)
bn_stats_f32(const float* __restrict__ x, const float* __restrict__ gamma,
             const float* __restrict__ beta, float* sA, float* sB,
             int HW, int B, size_t batchStride) {
  int c = blockIdx.x;
  float s = 0.f, q = 0.f;
  for (int b = 0; b < B; b++) {
    const float* p = x + (size_t)b * batchStride + (size_t)c * HW;
    for (int i = threadIdx.x; i < HW; i += 256) {
      float v = p[i]; s += v; q += v * v;
    }
  }
  __shared__ float rs[256], rq[256];
  int t = threadIdx.x; rs[t] = s; rq[t] = q; __syncthreads();
  for (int o = 128; o > 0; o >>= 1) {
    if (t < o) { rs[t] += rs[t + o]; rq[t] += rq[t + o]; }
    __syncthreads();
  }
  if (t == 0) {
    float N = (float)(B * HW);
    float m = rs[0] / N;
    float var = rq[0] / N - m * m;
    float rstd = rsqrtf(var + 1e-5f);
    float a = gamma[c] * rstd;
    sA[c] = a; sB[c] = beta[c] - m * a;
  }
}

// same but input is bf16 NHWC [B][HW][C]
__global__ void __launch_bounds__(256)
bn_stats_bf16(const unsigned short* __restrict__ x, const float* __restrict__ gamma,
              const float* __restrict__ beta, float* sA, float* sB,
              int HW, int B, int C) {
  int c = blockIdx.x;
  float s = 0.f, q = 0.f;
  for (int b = 0; b < B; b++) {
    const unsigned short* p = x + (size_t)b * HW * C + c;
    for (int i = threadIdx.x; i < HW; i += 256) {
      float v = bf2f(p[(size_t)i * C]); s += v; q += v * v;
    }
  }
  __shared__ float rs[256], rq[256];
  int t = threadIdx.x; rs[t] = s; rq[t] = q; __syncthreads();
  for (int o = 128; o > 0; o >>= 1) {
    if (t < o) { rs[t] += rs[t + o]; rq[t] += rq[t + o]; }
    __syncthreads();
  }
  if (t == 0) {
    float N = (float)(B * HW);
    float m = rs[0] / N;
    float var = rq[0] / N - m * m;
    float rstd = rsqrtf(var + 1e-5f);
    float a = gamma[c] * rstd;
    sA[c] = a; sB[c] = beta[c] - m * a;
  }
}

// NCHW f32 -> NHWC bf16 transpose (per batch), 32x32 LDS tiles
__global__ void __launch_bounds__(256)
transpose_cn(const float* __restrict__ in, unsigned short* __restrict__ out, int C, int N) {
  __shared__ float t[32][33];
  int b = blockIdx.z;
  int n0 = blockIdx.x * 32, c0 = blockIdx.y * 32;
  int tx = threadIdx.x & 31, ty = threadIdx.x >> 5;
  const float* ip = in + (size_t)b * C * N;
  unsigned short* op = out + (size_t)b * N * C;
  for (int r = ty; r < 32; r += 8) t[r][tx] = ip[(size_t)(c0 + r) * N + n0 + tx];
  __syncthreads();
  for (int r = ty; r < 32; r += 8) op[(size_t)(n0 + r) * C + c0 + tx] = f2bf(t[tx][r]);
}

// Fold BN2 into map_qv weights: W'[m][k] = W[m][k]*s2[k] (bf16), bias[m] = sum_k W[m][k]*t2[k]
__global__ void __launch_bounds__(256)
fold_kv(const float* __restrict__ Wsrc, const float* __restrict__ s2,
        const float* __restrict__ t2, unsigned short* __restrict__ Wb,
        float* __restrict__ bias) {
  int m = blockIdx.x, k = threadIdx.x;
  float w = Wsrc[(size_t)m * 256 + k];
  Wb[(size_t)m * 256 + k] = f2bf(w * s2[k]);
  __shared__ float red[256];
  red[k] = w * t2[k]; __syncthreads();
  for (int o = 128; o > 0; o >>= 1) { if (k < o) red[k] += red[k + o]; __syncthreads(); }
  if (k == 0) bias[m] = red[0];
}

// ---------------------------------------------------------------------------
// depthwise 3x3, stride 1, pad 1 on 64x64, fused per-channel affine on input.
// input NCHW f32 [B][256][64][64] -> output NHWC bf16 [B][4096][256]
// block = (y-row, 32-channel chunk, batch); 8 waves, lane == channel
// ---------------------------------------------------------------------------
__global__ void __launch_bounds__(256)
dw3x3(const float* __restrict__ in, const float* __restrict__ w9,
      const float* __restrict__ sA, const float* __restrict__ sB,
      unsigned short* __restrict__ out) {
  const int y = blockIdx.x;
  const int cb = blockIdx.y * 32;
  const int b = blockIdx.z;
  const int lane = threadIdx.x & 31, wv = threadIdx.x >> 5;
  __shared__ float tile[32 * 205];   // [c][3 rows][68 cols], padded vs bank conflicts
  for (int idx = threadIdx.x; idx < 32 * 3 * 68; idx += 256) {
    int c = idx / (3 * 68); int rem = idx % (3 * 68);
    int ky = rem / 68; int xx = rem % 68;
    float v = 0.f;
    int xi = xx - 1, yi = y + ky - 1;
    if (xx < 66 && xi >= 0 && xi < 64 && yi >= 0 && yi < 64) {
      int cg = cb + c;
      v = in[(((size_t)b * 256 + cg) * 64 + yi) * 64 + xi] * sA[cg] + sB[cg];
    }
    tile[c * 205 + ky * 68 + xx] = v;
  }
  __syncthreads();
  const int c = lane, cg = cb + c;
  float w[9];
#pragma unroll
  for (int t = 0; t < 9; t++) w[t] = w9[cg * 9 + t];
#pragma unroll
  for (int xo = 0; xo < 8; xo++) {
    int x = wv * 8 + xo;
    float a = 0.f;
#pragma unroll
    for (int ky = 0; ky < 3; ky++)
#pragma unroll
      for (int kx = 0; kx < 3; kx++)
        a += w[ky * 3 + kx] * tile[c * 205 + ky * 68 + x + kx];
    out[(((size_t)b * 4096) + y * 64 + x) * 256 + cg] = f2bf(a);  // 32 lanes contiguous
  }
}

// ---------------------------------------------------------------------------
// Pointwise (1x1) conv as WMMA GEMM on bf16 activations, computing Out^T:
//   Out[m][n] = sum_k W[m][k] * X[n][k]   (X bf16 NHWC -> k contiguous)
// No LDS, no barriers: A- and B-fragments are direct 16B global loads that
// use contiguous k = 16*(lane/16) + s for both A(16x32) and B(32x16).
// block tile 128n x 128m, 8 waves as 4(n) x 2(m), wave tile 32n x 64m.
// mode: 0 = bf16 NHWC out, 1 = Q bf16 (scaled, head scatter),
//       2 = KV bf16 (+bias; K head scatter, V transposed [d][n]), 3 = f32 NCHW out
// ---------------------------------------------------------------------------
__global__ void __launch_bounds__(256)
gemm_pw(const unsigned short* __restrict__ W, const unsigned short* __restrict__ X,
        int K, int Mtot, int mode,
        float* __restrict__ outF, unsigned short* __restrict__ outU,
        unsigned short* __restrict__ outU2, const float* __restrict__ bias) {
  const int b = blockIdx.z;
  const int nb = blockIdx.x * 128;
  const int mbb = blockIdx.y * 128;
  const int lane = threadIdx.x & 31;
  const int wv = threadIdx.x >> 5;
  const int ng = wv >> 1;   // 0..3 -> 32 n each
  const int mg = wv & 1;    // 0..1 -> 64 m each

  FragC acc[2][4];
#pragma unroll
  for (int i = 0; i < 2; i++)
#pragma unroll
    for (int j = 0; j < 4; j++)
#pragma unroll
      for (int r = 0; r < 8; r++) acc[i][j].f[r] = 0.f;

  const unsigned short* Xb = X + (size_t)b * 4096 * K;
  const int klo = (lane >> 4) * 8;    // unused (A-frag also uses kb2)
  const int kb2 = (lane >> 4) * 16;   // B-frag K offset within chunk

  for (int kc = 0; kc < K; kc += 32) {
    Frag16 Af[2];
#pragma unroll
    for (int nt = 0; nt < 2; nt++) {
      int nr = nb + ng * 32 + nt * 16 + (lane & 15);
      const unsigned short* ap = Xb + (size_t)nr * K + kc;
      Af[nt].q[0] = *(const uint4*)(ap + kb2);        // elems 0..7  : K = kb2..kb2+7
      Af[nt].q[1] = *(const uint4*)(ap + kb2 + 8);    // elems 8..15 : K = kb2+8..kb2+15
    }
    Frag16 Bf[4];
#pragma unroll
    for (int mt = 0; mt < 4; mt++) {
      int m = mbb + mg * 64 + mt * 16 + (lane & 15);
      const uint4* p = (const uint4*)(W + (size_t)m * K + kc + kb2);  // 16 contiguous
      Bf[mt].q[0] = p[0]; Bf[mt].q[1] = p[1];
    }
    if (kc + 32 < K) {  // global_prefetch_b8 of next chunks
      __builtin_prefetch(Xb + (size_t)(nb + ng * 32 + (lane & 15)) * K + kc + 32, 0, 1);
      __builtin_prefetch(W + (size_t)(mbb + mg * 64 + (lane & 15)) * K + kc + 32, 0, 1);
    }
#pragma unroll
    for (int nt = 0; nt < 2; nt++)
#pragma unroll
      for (int mt = 0; mt < 4; mt++)
        acc[nt][mt].v = WMMA_BF16(Af[nt].v, Bf[mt].v, acc[nt][mt].v);
  }

  // epilogue: C-frag lane holds col m = L%16(+tile), rows n = (L/16)*8 + r
#pragma unroll
  for (int nt = 0; nt < 2; nt++) {
#pragma unroll
    for (int mt = 0; mt < 4; mt++) {
      int m = mbb + mg * 64 + mt * 16 + (lane & 15);
      int n0 = nb + ng * 32 + nt * 16 + ((lane >> 4) * 8);
      FragC c = acc[nt][mt];
      if (mode == 0) {                       // bf16 NHWC
        unsigned short* o = outU + ((size_t)b * 4096 + n0) * Mtot + m;
#pragma unroll
        for (int r = 0; r < 8; r++) o[(size_t)r * Mtot] = f2bf(c.f[r]);
      } else if (mode == 1) {                // Q bf16, scaled, [b][h][n][d]
        int h = m & 3, d = m >> 2;
        unsigned short* o = outU + (((size_t)b * 4 + h) * 4096 + n0) * 64 + d;
#pragma unroll
        for (int r = 0; r < 8; r++) o[(size_t)r * 64] = f2bf(c.f[r] * 0.125f);
      } else if (mode == 2) {                // K / V^T with bias
        float bm = bias[m];
        if (m < 256) {
          int h = m & 3, d = m >> 2;
          unsigned short* o = outU + (((size_t)b * 4 + h) * 4096 + n0) * 64 + d;
#pragma unroll
          for (int r = 0; r < 8; r++) o[(size_t)r * 64] = f2bf(c.f[r] + bm);
        } else {
          int cc = m - 256, h = cc & 3, d = cc >> 2;
          Pack8 pk;
#pragma unroll
          for (int r = 0; r < 8; r++) pk.us[r] = f2bf(c.f[r] + bm);
          *(uint4*)(outU2 + (((size_t)b * 4 + h) * 64 + d) * 4096 + n0) = pk.q;
        }
      } else {                               // f32 NCHW (final output)
        float* o = outF + ((size_t)b * 256 + m) * 4096 + n0;
        *(float4*)o = make_float4(c.f[0], c.f[1], c.f[2], c.f[3]);
        *(float4*)(o + 4) = make_float4(c.f[4], c.f[5], c.f[6], c.f[7]);
      }
    }
  }
}

// ---------------------------------------------------------------------------
// Flash attention, transposed formulation.
//   S^T = K * Q^T  (softmax per C-frag column == per query, lane-local)
//   O^T += V^T * P^T
// Q bf16 [b][h][4096][64] (pre-scaled), K bf16 [b][h][4096][64], V^T bf16 [b][h][64][4096]
// out: attn_out NCHW f32 [b][256][4096], channel c = d*4 + h
// block = 64 queries; 8 waves: (jg = wv>>1) x (ig = wv&1); Bc = 64 keys/iter.
// ---------------------------------------------------------------------------
__global__ void __launch_bounds__(256)
attn_kernel(const unsigned short* __restrict__ Q, const unsigned short* __restrict__ K,
            const unsigned short* __restrict__ V, float* __restrict__ attn_out) {
  const int qblk = blockIdx.x;   // 0..63
  const int bh = blockIdx.y;     // b*4 + h
  const int tid = threadIdx.x;
  const int lane = tid & 31;
  const int wv = tid >> 5;
  const int jg = wv >> 1;        // S phase: key group / O phase: d group
  const int ig = wv & 1;         // query group (both phases)

  __shared__ __align__(16) unsigned short Pbuf[2][4][32][16];  // P^T in B-frag layout
  __shared__ float partmax[4][64], partsum[4][64];
  __shared__ float m_run[64], l_run[64], salpha[64];

  if (tid < 64) { m_run[tid] = -3.0e38f; l_run[tid] = 0.f; }
  __syncthreads();

  // loop-invariant Q^T B-fragments (kept in VGPRs)
  const size_t baseQ = ((size_t)bh * 4096 + (size_t)qblk * 64) * 64;
  Frag16 Qf[2][2];
#pragma unroll
  for (int kc = 0; kc < 2; kc++)
#pragma unroll
    for (int it = 0; it < 2; it++) {
      int iL = ig * 32 + it * 16 + (lane & 15);
      const uint4* p = (const uint4*)(Q + baseQ + (size_t)iL * 64 + kc * 32 + ((lane >> 4) * 16));
      Qf[kc][it].q[0] = p[0]; Qf[kc][it].q[1] = p[1];
    }

  FragC Oacc[2];
#pragma unroll
  for (int it = 0; it < 2; it++)
#pragma unroll
    for (int r = 0; r < 8; r++) Oacc[it].f[r] = 0.f;

  const size_t baseK = (size_t)bh * 4096 * 64;
  const size_t baseV = (size_t)bh * 64 * 4096;

  for (int jt = 0; jt < 64; jt++) {
    const int jbase = jt * 64;

    // ---- S^T = K x Q^T (frag rows = keys, cols = queries)
    FragC S[2];
#pragma unroll
    for (int it = 0; it < 2; it++)
#pragma unroll
      for (int r = 0; r < 8; r++) S[it].f[r] = 0.f;
#pragma unroll
    for (int kc = 0; kc < 2; kc++) {
      Frag16 Ak;
      int j = jbase + jg * 16 + (lane & 15);
      const unsigned short* kp = K + baseK + (size_t)j * 64 + kc * 32 + ((lane >> 4) * 16);
      Ak.q[0] = *(const uint4*)kp;
      Ak.q[1] = *(const uint4*)(kp + 8);
#pragma unroll
      for (int it = 0; it < 2; it++)
        S[it].v = WMMA_BF16(Ak.v, Qf[kc][it].v, S[it].v);
    }

    // ---- per-query tile max (in-lane + one shfl_xor(16))
#pragma unroll
    for (int it = 0; it < 2; it++) {
      float m8 = S[it].f[0];
#pragma unroll
      for (int r = 1; r < 8; r++) m8 = fmaxf(m8, S[it].f[r]);
      float m16 = fmaxf(m8, __shfl_xor(m8, 16));
      if (lane < 16) partmax[jg][ig * 32 + it * 16 + lane] = m16;
    }
    __syncthreads();
    if (tid < 64) {
      float mt = fmaxf(fmaxf(partmax[0][tid], partmax[1][tid]),
                       fmaxf(partmax[2][tid], partmax[3][tid]));
      float mo = m_run[tid];
      float mn = fmaxf(mo, mt);
      salpha[tid] = __expf(mo - mn);
      m_run[tid] = mn;
    }
    __syncthreads();

    // ---- exponentiate, partial sums, pack P^T into B-frag-layout LDS
#pragma unroll
    for (int it = 0; it < 2; it++) {
      int iL = ig * 32 + it * 16 + (lane & 15);
      float mn = m_run[iL];
      Pack8 pk; float s8 = 0.f;
#pragma unroll
      for (int r = 0; r < 8; r++) {
        float p = __expf(S[it].f[r] - mn);
        s8 += p; pk.us[r] = f2bf(p);
      }
      float s16 = s8 + __shfl_xor(s8, 16);
      if (lane < 16) partsum[jg][ig * 32 + it * 16 + lane] = s16;
      int kc = jg >> 1;
      int Lc = (lane & 15) + ((jg & 1) << 4);
      *(uint4*)&Pbuf[kc][ig * 2 + it][Lc][(lane >> 4) * 8] = pk.q;
    }
    // rescale running O by alpha (per-lane scalar: column = query)
#pragma unroll
    for (int it = 0; it < 2; it++) {
      float a = salpha[ig * 32 + it * 16 + (lane & 15)];
#pragma unroll
      for (int r = 0; r < 8; r++) Oacc[it].f[r] *= a;
    }
    __syncthreads();
    if (tid < 64) {
      float ts = partsum[0][tid] + partsum[1][tid] + partsum[2][tid] + partsum[3][tid];
      l_run[tid] = l_run[tid] * salpha[tid] + ts;
    }

    // ---- O^T += V^T x P^T  (wave: d group jg, query group ig)
#pragma unroll
    for (int kc = 0; kc < 2; kc++) {
      Frag16 Av;
      int d = jg * 16 + (lane & 15);
      const unsigned short* vp = V + baseV + (size_t)d * 4096 + jbase + kc * 32 + ((lane >> 4) * 16);
      Av.q[0] = *(const uint4*)vp;
      Av.q[1] = *(const uint4*)(vp + 8);
#pragma unroll
      for (int it = 0; it < 2; it++) {
        Frag16 Pf;
        Pf.q[0] = *(const uint4*)&Pbuf[kc][ig * 2 + it][lane][0];
        Pf.q[1] = *(const uint4*)&Pbuf[kc][ig * 2 + it][lane][8];
        Oacc[it].v = WMMA_BF16(Av.v, Pf.v, Oacc[it].v);
      }
    }
    __syncthreads();
  }

  // ---- normalize + store to NCHW, c = d*4 + h
  const int b = bh >> 2, h = bh & 3;
#pragma unroll
  for (int it = 0; it < 2; it++) {
    int iL = ig * 32 + it * 16 + (lane & 15);
    float l = l_run[iL];
    float rl = (l > 0.f) ? 1.f / l : 0.f;
    int iG = qblk * 64 + iL;
#pragma unroll
    for (int r = 0; r < 8; r++) {
      int d = jg * 16 + ((lane >> 4) * 8) + r;
      int c = d * 4 + h;
      attn_out[((size_t)b * 256 + c) * 4096 + iG] = Oacc[it].f[r] * rl;
    }
  }
}

// ---------------------------------------------------------------------------
// host side
// ---------------------------------------------------------------------------
extern "C" void kernel_launch(void* const* d_in, const int* in_sizes, int n_in,
                              void* d_out, int out_size, void* d_ws, size_t ws_size,
                              hipStream_t stream) {
  (void)in_sizes; (void)n_in; (void)out_size; (void)ws_size;
  const float* feat   = (const float*)d_in[0];   // [2,256,64,64]
  const float* smap   = (const float*)d_in[1];   // [2,160,64,64]
  const float* mapw   = (const float*)d_in[2];   // [256,160]
  const float* gamma1 = (const float*)d_in[3];
  const float* beta1  = (const float*)d_in[4];
  const float* gamma2 = (const float*)d_in[5];
  const float* beta2  = (const float*)d_in[6];
  const float* qv_dw  = (const float*)d_in[7];   // [256,1,3,3]
  const float* qv_pw  = (const float*)d_in[8];   // [512,256] (rows 0..255 = Q)
  const float* kv_w   = (const float*)d_in[9];   // [512,256]
  const float* out_dw = (const float*)d_in[10];  // [256,1,3,3]
  const float* out_pw = (const float*)d_in[11];  // [256,256]
  float* out = (float*)d_out;

  char* ws = (char*)d_ws;
  size_t off = 0;
  auto alloc = [&](size_t bytes) -> char* {
    char* p = ws + off;
    off += (bytes + 255) & ~(size_t)255;
    return p;
  };
  float* sA1   = (float*)alloc(256 * 4);
  float* sB1   = (float*)alloc(256 * 4);
  float* s2    = (float*)alloc(256 * 4);
  float* t2    = (float*)alloc(256 * 4);
  float* ones  = (float*)alloc(256 * 4);
  float* zeros = (float*)alloc(256 * 4);
  float* bias2 = (float*)alloc(512 * 4);
  unsigned short* wq_bf   = (unsigned short*)alloc(256 * 256 * 2);
  unsigned short* wout_bf = (unsigned short*)alloc(256 * 256 * 2);
  unsigned short* wkv_bf  = (unsigned short*)alloc(512 * 256 * 2);
  unsigned short* wmap_bf = (unsigned short*)alloc(256 * 160 * 2);
  unsigned short* semT    = (unsigned short*)alloc((size_t)2 * 4096 * 160 * 2);  // bf16 NHWC
  unsigned short* sm_pre  = (unsigned short*)alloc((size_t)2 * 4096 * 256 * 2);  // bf16 NHWC
  unsigned short* dwx     = (unsigned short*)alloc((size_t)2 * 4096 * 256 * 2);  // bf16 NHWC
  unsigned short* qbuf = (unsigned short*)alloc((size_t)8 * 4096 * 64 * 2);
  unsigned short* kbuf = (unsigned short*)alloc((size_t)8 * 4096 * 64 * 2);
  unsigned short* vbuf = (unsigned short*)alloc((size_t)8 * 64 * 4096 * 2);
  float* attn_out = (float*)alloc((size_t)2 * 256 * 4096 * 4);
  unsigned short* dwout = (unsigned short*)alloc((size_t)2 * 4096 * 256 * 2);    // bf16 NHWC

  init_consts<<<1, 256, 0, stream>>>(ones, zeros);
  cvt_f32_bf16<<<256, 256, 0, stream>>>(qv_pw, wq_bf, 256 * 256);
  cvt_f32_bf16<<<256, 256, 0, stream>>>(out_pw, wout_bf, 256 * 256);
  cvt_f32_bf16<<<160, 256, 0, stream>>>(mapw, wmap_bf, 256 * 160);

  // BN1 stats over NCHW feat -> folded scale/shift
  bn_stats_f32<<<256, 256, 0, stream>>>(feat, gamma1, beta1, sA1, sB1,
                                        4096, 2, (size_t)256 * 4096);
  // semantic map to NHWC bf16
  transpose_cn<<<dim3(128, 5, 2), 256, 0, stream>>>(smap, semT, 160, 4096);
  // map projection (1x1): sm_pre = semT x map_proj_w^T  (bf16 NHWC out)
  gemm_pw<<<dim3(32, 2, 2), 256, 0, stream>>>(wmap_bf, semT, 160, 256, 0,
                                              nullptr, sm_pre, nullptr, nullptr);
  // BN2 stats over bf16 NHWC sm_pre -> fold factors
  bn_stats_bf16<<<256, 256, 0, stream>>>(sm_pre, gamma2, beta2, s2, t2, 4096, 2, 256);
  // fold BN2 into map_qv weights + bias
  fold_kv<<<512, 256, 0, stream>>>(kv_w, s2, t2, wkv_bf, bias2);
  // BN1 + depthwise 3x3 on feat -> NHWC bf16
  dw3x3<<<dim3(64, 8, 2), 256, 0, stream>>>(feat, qv_dw, sA1, sB1, dwx);
  // Q = dwx x Wq^T, scaled by 1/8, bf16 [b][h][n][d]
  gemm_pw<<<dim3(32, 2, 2), 256, 0, stream>>>(wq_bf, dwx, 256, 256, 1,
                                              nullptr, qbuf, nullptr, nullptr);
  // K / V^T = sm_pre x Wkv'^T + bias
  gemm_pw<<<dim3(32, 4, 2), 256, 0, stream>>>(wkv_bf, sm_pre, 256, 512, 2,
                                              nullptr, kbuf, vbuf, bias2);
  // flash attention -> attn_out NCHW f32
  attn_kernel<<<dim3(64, 8), 256, 0, stream>>>(qbuf, kbuf, vbuf, attn_out);
  // output depthwise 3x3 (no norm) -> NHWC bf16
  dw3x3<<<dim3(64, 8, 2), 256, 0, stream>>>(attn_out, out_dw, ones, zeros, dwout);
  // final pointwise -> d_out NCHW f32
  gemm_pw<<<dim3(32, 2, 2), 256, 0, stream>>>(wout_bf, dwout, 256, 256, 3,
                                              out, nullptr, nullptr, nullptr);
}
